// GumbelVQ_72576357368195
// MI455X (gfx1250) — compile-verified, matrix operand order checked by reading, not run on hardware
//
#include <hip/hip_runtime.h>
#include <hip/hip_bf16.h>

typedef __attribute__((ext_vector_type(16))) _Float16 v16h;
typedef __attribute__((ext_vector_type(8)))  _Float16 v8h;
typedef __attribute__((ext_vector_type(2)))  _Float16 v2h;
typedef __attribute__((ext_vector_type(8)))  float    v8f;

#define EMB_K 256
#define EMB_D 32
#define M_TOT (16 * 16384)
#define ROWS_PER_WAVE 16
#define WAVES_PER_BLOCK 4
#define ROWS_PER_BLOCK (ROWS_PER_WAVE * WAVES_PER_BLOCK)
#define T_VAL 1.0f
#define TR_VAL (1.0f / 3.0f)   // (T - Tmin)/(Tinit - Tmin) = 0.5/1.5

#define LN2_F   0.6931471805599453f
#define LOG2E_F 1.4426950408889634f

// d_out layout (flattened tuple, fp32):
//   [0, 8388608)            quantized [16,16384,32]
//   8388608                 commitment_loss
//   [8388609, 8650753)      indices [16,16384] (as float)
//   8650753                 perplexity
#define OFF_CLOSS 8388608
#define OFF_IDX   8388609
#define OFF_PERP  8650753

// Raw-hardware transcendentals (inputs here are never denormal, so the
// safe ldexp/cndmask wrappers around v_log_f32 / v_exp_f32 are dead weight).
__device__ __forceinline__ float fast_ln(float v)   { return __builtin_amdgcn_logf(v) * LN2_F; }
__device__ __forceinline__ float fast_exp2(float v) { return __builtin_amdgcn_exp2f(v); }
__device__ __forceinline__ float fast_rcp(float v)  { return __builtin_amdgcn_rcpf(v); }

__global__ void vq_init_ws(float* ws) {
    int t = threadIdx.x + blockIdx.x * blockDim.x;
    if (t < EMB_K + 1) ws[t] = 0.0f;
}

__global__ __launch_bounds__(128)
void vq_main(const float* __restrict__ x, const float* __restrict__ emb,
             const float* __restrict__ gum, float* __restrict__ outq,
             float* __restrict__ outidx, float* __restrict__ ws) {
    __shared__ _Float16 s_embf16[EMB_K * EMB_D];            // 16 KB  emb[k][d] f16
    __shared__ _Float16 s_embT16[EMB_D * EMB_K];            // 16 KB  emb[d][k] f16
    __shared__ float    s_enorm[EMB_K];                     //  1 KB  ||e_k||^2
    __shared__ alignas(16) _Float16 s_z[WAVES_PER_BLOCK * ROWS_PER_WAVE * EMB_K]; // 32 KB
    __shared__ alignas(16) float s_hard[WAVES_PER_BLOCK * ROWS_PER_WAVE * EMB_D]; // 8 KB
    __shared__ float    s_invZ[WAVES_PER_BLOCK * ROWS_PER_WAVE];
    __shared__ float    s_probs[EMB_K];
    __shared__ float    s_closs;

    const int tid  = threadIdx.x;
    const int wave = tid >> 5;
    const int lane = tid & 31;
    const int n    = lane & 15;   // col / row-in-tile index
    const int half = lane >> 4;   // lane group 0/1

    // ---- stage embedding into LDS (both layouts) + norms, zero accumulators ----
    for (int i = tid; i < EMB_K * EMB_D; i += 128) {
        int k = i >> 5, d = i & 31;
        float e = emb[i];
        s_embf16[i] = (_Float16)e;
        s_embT16[d * EMB_K + k] = (_Float16)e;
    }
    for (int k = tid; k < EMB_K; k += 128) {
        float s = 0.0f;
        for (int d = 0; d < EMB_D; ++d) { float e = emb[k * EMB_D + d]; s += e * e; }
        s_enorm[k] = s;
        s_probs[k] = 0.0f;
    }
    if (tid == 0) s_closs = 0.0f;
    __syncthreads();

    const int rowBase = blockIdx.x * ROWS_PER_BLOCK + wave * ROWS_PER_WAVE;
    _Float16* zb = s_z + wave * ROWS_PER_WAVE * EMB_K;

    // ---- A fragment: x tile 16x32 in f16 (ISA 16-bit A 16x32 layout) ----
    v16h aX;
    {
        int rg = rowBase + n;
        #pragma unroll
        for (int v = 0; v < 8; ++v) {
            int db = ((v < 4) ? 0 : 16) + half * 8 + (v & 3) * 2;
            float2 p = *(const float2*)(x + (size_t)rg * EMB_D + db);
            aX[2 * v]     = (_Float16)p.x;
            aX[2 * v + 1] = (_Float16)p.y;
        }
    }

    const float invT = 1.0f / T_VAL;

    // ================= phase 1: logits + gumbel -> z*log2e in LDS =================
    for (int c = 0; c < 16; ++c) {
        int code = c * 16 + n;
        v16h bE;   // B = emb^T chunk: B[d][n] = emb[c*16+n][d], K=32, N=16
        #pragma unroll
        for (int v = 0; v < 8; ++v) {
            int db = ((v < 4) ? 0 : 16) + half * 8 + (v & 3) * 2;
            v2h p = *(const v2h*)(s_embf16 + code * EMB_D + db);
            bE[2 * v] = p[0]; bE[2 * v + 1] = p[1];
        }
        v8f acc = {};
        acc = __builtin_amdgcn_wmma_f32_16x16x32_f16(false, aX, false, bE,
                                                     (short)0, acc, false, false);
        if (c < 15)
            __builtin_prefetch(gum + (size_t)(rowBase + half * 8) * EMB_K + (c + 1) * 16 + n, 0, 1);
        float en = s_enorm[code];
        #pragma unroll
        for (int r = 0; r < 8; ++r) {
            int rl = r + half * 8;
            float u = gum[(size_t)(rowBase + rl) * EMB_K + code];
            float g = -fast_ln(-fast_ln(u + 1e-20f) + 1e-20f);
            float l = (2.0f * acc[r] - en) * invT;       // ||x||^2 term cancels
            // store z scaled by log2(e): exp(z) == exp2(z*log2e); argmax preserved
            zb[rl * EMB_K + code] = (_Float16)((l + g) * invT * LOG2E_F);
        }
    }

    // ========== phase 2: softmax / argmax per row (2 lanes per row) ==========
    const int row = n;
    const int k0  = half * 128;
    _Float16* zr = zb + row * EMB_K + k0;   // 256B-aligned -> v8h ok
    v8h* zr8 = (v8h*)zr;

    float mx = -3.0e38f; int ai = k0;
    for (int k8 = 0; k8 < 16; ++k8) {       // 16x ds_load_b128 instead of 128x u16
        v8h zz = zr8[k8];
        #pragma unroll
        for (int j = 0; j < 8; ++j) {
            float zv = (float)zz[j];
            if (zv > mx) { mx = zv; ai = k0 + k8 * 8 + j; }
        }
    }
    {
        float mo = __shfl_xor(mx, 16, 32);
        int   io = __shfl_xor(ai, 16, 32);
        if (mo > mx || (mo == mx && io < ai)) { mx = mo; ai = io; }
    }
    float Zs = 0.0f;
    for (int k8 = 0; k8 < 16; ++k8) {
        v8h zz = zr8[k8];
        v8h ee;
        #pragma unroll
        for (int j = 0; j < 8; ++j) {
            float e = fast_exp2((float)zz[j] - mx);   // bare v_exp_f32
            ee[j] = (_Float16)e;
            Zs += e;
        }
        zr8[k8] = ee;                                  // one ds_store_b128
    }
    Zs += __shfl_xor(Zs, 16, 32);
    float invZ = fast_rcp(Zs);

    for (int k8 = 0; k8 < 16; ++k8) {
        v8h ee = zr8[k8];
        #pragma unroll
        for (int j = 0; j < 8; ++j)
            atomicAdd(&s_probs[k0 + k8 * 8 + j], (float)ee[j] * invZ);
    }

    if (half == 0) {
        s_invZ[wave * 16 + row] = invZ;
        int rg = rowBase + row;
        outidx[rg] = (float)ai;
        float cl = 0.0f;
        #pragma unroll
        for (int d4 = 0; d4 < 8; ++d4) {               // 128-bit loads/stores
            float4 h  = *(const float4*)(emb + ai * EMB_D + d4 * 4);
            float4 xv = *(const float4*)(x + (size_t)rg * EMB_D + d4 * 4);
            float4 hs = make_float4((1.0f - TR_VAL) * h.x, (1.0f - TR_VAL) * h.y,
                                    (1.0f - TR_VAL) * h.z, (1.0f - TR_VAL) * h.w);
            *(float4*)(s_hard + (wave * 16 + row) * EMB_D + d4 * 4) = hs;
            float dx = h.x - xv.x, dy = h.y - xv.y, dz = h.z - xv.z, dw = h.w - xv.w;
            cl += dx * dx + dy * dy + dz * dz + dw * dw;
        }
        atomicAdd(&s_closs, cl);
    }

    // ====== phase 3: quantized = tr * (exp @ E) / Z + (1-tr) * hard ======
    #pragma unroll
    for (int dt = 0; dt < 2; ++dt) {
        v8f acc = {};
        for (int kc = 0; kc < 8; ++kc) {
            v16h aS, bS;
            #pragma unroll
            for (int v = 0; v < 8; ++v) {
                int kb = ((v < 4) ? 0 : 16) + half * 8 + (v & 3) * 2;
                v2h p = *(const v2h*)(zb + row * EMB_K + kc * 32 + kb);
                aS[2 * v] = p[0]; aS[2 * v + 1] = p[1];
                v2h q = *(const v2h*)(s_embT16 + (dt * 16 + n) * EMB_K + kc * 32 + kb);
                bS[2 * v] = q[0]; bS[2 * v + 1] = q[1];
            }
            acc = __builtin_amdgcn_wmma_f32_16x16x32_f16(false, aS, false, bS,
                                                         (short)0, acc, false, false);
        }
        #pragma unroll
        for (int r = 0; r < 8; ++r) {
            int rl = r + half * 8;
            float qv = TR_VAL * s_invZ[wave * 16 + rl] * acc[r]
                     + s_hard[(wave * 16 + rl) * EMB_D + dt * 16 + n];
            outq[(size_t)(rowBase + rl) * EMB_D + dt * 16 + n] = qv;
        }
    }

    // ---- block-level flush of accumulators ----
    __syncthreads();
    for (int i = tid; i < EMB_K; i += 128)
        atomicAdd(&ws[i], s_probs[i]);
    if (tid == 0) atomicAdd(&ws[EMB_K], s_closs);
}

__global__ void vq_final(const float* __restrict__ ws, float* __restrict__ out) {
    __shared__ float red[EMB_K];
    int t = threadIdx.x;
    float p = ws[t] * (1.0f / (float)M_TOT);
    red[t] = p * fast_ln(p + 1e-10f);
    __syncthreads();
    for (int s = EMB_K / 2; s > 0; s >>= 1) {
        if (t < s) red[t] += red[t + s];
        __syncthreads();
    }
    if (t == 0) {
        out[OFF_CLOSS] = ws[EMB_K] * (1.0f / (float)(M_TOT * EMB_D));
        out[OFF_PERP]  = fast_exp2(-red[0] * LOG2E_F);
    }
}

extern "C" void kernel_launch(void* const* d_in, const int* in_sizes, int n_in,
                              void* d_out, int out_size, void* d_ws, size_t ws_size,
                              hipStream_t stream) {
    const float* x   = (const float*)d_in[0];   // [16,16384,32]
    const float* emb = (const float*)d_in[1];   // [256,32]
    const float* gum = (const float*)d_in[2];   // [262144,256]
    float* out = (float*)d_out;
    float* ws  = (float*)d_ws;

    vq_init_ws<<<3, 128, 0, stream>>>(ws);
    vq_main<<<M_TOT / ROWS_PER_BLOCK, 128, 0, stream>>>(
        x, emb, gum, out, out + OFF_IDX, ws);
    vq_final<<<1, EMB_K, 0, stream>>>(ws, out);
}
